// MultiHeadSelfAttention_30442728194229
// MI455X (gfx1250) — matrix operand order repaired and checked by reading, not compiled, not run on hardware
//
#include <hip/hip_runtime.h>
#include <hip/hip_bf16.h>

typedef __bf16 bf16_t;
typedef bf16_t bf16x8  __attribute__((ext_vector_type(8)));
typedef bf16_t bf16x16 __attribute__((ext_vector_type(16)));
typedef float  f32x8   __attribute__((ext_vector_type(8)));
typedef unsigned int u32x4 __attribute__((ext_vector_type(4)));
typedef int i32x4 __attribute__((ext_vector_type(4)));
typedef int i32x8 __attribute__((ext_vector_type(8)));

#define DEVINL __device__ __forceinline__

static constexpr int BATCH = 8;
static constexpr int SEQ   = 1024;
static constexpr int DIM   = 1024;
static constexpr int NHEAD = 16;
static constexpr int HDIM  = 64;            // DIM / NHEAD
static constexpr int MTOT  = BATCH * SEQ;   // 8192

// GEMM pipeline stage: 64 K-elements per TDM tile.
// LDS row stride: 64 data elems + 8 pad elems (TDM pads 16B after each 128B row)
static constexpr int KSTEP = 64;
static constexpr int LPAD  = 72;

union BF16Frag {
  bf16x16 v;
  bf16x8  h[2];
};

DEVINL f32x8 wmma_bf16(bf16x16 a, bf16x16 b, f32x8 c) {
  // D = A(16x32) * B(32x16) + C, f32 accumulate
  return __builtin_amdgcn_wmma_f32_16x16x32_bf16(false, a, false, b, (short)0, c,
                                                 false, false);
}

DEVINL unsigned lds_off(const void* p) {
  // generic pointer to LDS: low 32 bits are the LDS byte address
  return (unsigned)(unsigned long long)(uintptr_t)p;
}

// TDM: 2-D tile load (rows x 64 bf16) from a row-major [*, DIM] bf16 matrix
// into LDS with 16B padding after every 128B row (LDS row stride = 144B).
DEVINL void tdm_load_tile(unsigned ldsaddr, const bf16_t* gptr, int rows) {
  const unsigned long long ga = (unsigned long long)(uintptr_t)gptr;
  u32x4 g0;
  g0[0] = 1u;                                   // count=1, user mode, no gather
  g0[1] = ldsaddr;                              // lds_addr (bytes)
  g0[2] = (unsigned)ga;                         // global_addr[31:0]
  g0[3] = (unsigned)(ga >> 32) | (2u << 30);    // global_addr[56:32] | type=2

  i32x8 g1;
  // data_size=1 (2B) | pad_enable | pad_interval=4 (32 DW=128B) | pad_amount=3 (4 DW=16B)
  g1[0] = 0x07110000;
  g1[1] = 0x04000000;            // tensor_dim0[15:0]=1024 in bits[63:48]
  g1[2] = 0x04000000;            // tensor_dim1[15:0]=1024 in bits[95:80]
  g1[3] = KSTEP << 16;           // tile_dim0=64 in bits[127:112]
  g1[4] = rows;                  // tile_dim1=rows; tile_dim2=0
  g1[5] = 1024;                  // tensor_dim0_stride[31:0]
  g1[6] = 0;
  g1[7] = 0;

  i32x4 z{};
#if __clang_major__ >= 23
  i32x8 z8{};
  __builtin_amdgcn_tensor_load_to_lds(g0, g1, z, z, z8, 0);
#else
  __builtin_amdgcn_tensor_load_to_lds(g0, g1, z, z, 0);
#endif
}

// ---------------------------------------------------------------------------
// fp32 -> bf16 bulk convert (n must be a multiple of 8; all our sizes are)
// ---------------------------------------------------------------------------
__global__ __launch_bounds__(256) void cvt_f32_to_bf16(const float* __restrict__ src,
                                                       bf16_t* __restrict__ dst,
                                                       int n) {
  int i = (blockIdx.x * 256 + threadIdx.x) * 8;
  if (i + 8 <= n) {
    float4 f0 = *(const float4*)(src + i);
    float4 f1 = *(const float4*)(src + i + 4);
    bf16x8 o;
    o[0] = (bf16_t)f0.x; o[1] = (bf16_t)f0.y; o[2] = (bf16_t)f0.z; o[3] = (bf16_t)f0.w;
    o[4] = (bf16_t)f1.x; o[5] = (bf16_t)f1.y; o[6] = (bf16_t)f1.z; o[7] = (bf16_t)f1.w;
    *(bf16x8*)(dst + i) = o;
  }
}

// ---------------------------------------------------------------------------
// C = A(M,K) @ W(N,K)^T  with M=8192, N=K=1024, bf16 in / f32 accumulate.
// TDM double-buffered LDS staging: A tile 128x64 (wave0 DMA), B tile 64x64
// (wave1 DMA) per stage; 8 WMMAs per wave per stage.
// MODE 0: bf16 out, row-major [M, DIM]
// MODE 1: bf16 out, per-batch transposed: out[b, n, t]  (for V^T)
// MODE 2: f32  out, row-major [M, DIM]
// Block = 8 waves (256 thr), block tile 128(M) x 64(N), wave tile 32x32.
// ---------------------------------------------------------------------------
template <int MODE>
__global__ __launch_bounds__(256) void gemm_xwt(const bf16_t* __restrict__ A,
                                                const bf16_t* __restrict__ W,
                                                bf16_t* __restrict__ outb,
                                                float* __restrict__ outf) {
  __shared__ __align__(16) bf16_t lA[2][128][LPAD];   // 36 KB
  __shared__ __align__(16) bf16_t lB[2][64][LPAD];    // 18 KB

  const int lane = threadIdx.x & 31;
  const int wave = threadIdx.x >> 5;
  const int lm   = lane & 15;
  const int lh   = lane >> 4;

  const int mt = (wave >> 1) * 32;           // local tile row (within 128)
  const int nt = (wave & 1) * 32;            // local tile col (within 64)
  const int m0 = blockIdx.x * 128;
  const int n0 = blockIdx.y * 64;

  const bf16_t* gA = A + (size_t)m0 * DIM;   // 128 rows of A
  const bf16_t* gW = W + (size_t)n0 * DIM;   // 64 rows of W

  if (wave == 0) tdm_load_tile(lds_off(&lA[0][0][0]), gA, 128);
  if (wave == 1) tdm_load_tile(lds_off(&lB[0][0][0]), gW, 64);

  const int akoff = lh * 8;    // A: lanes 16-31 start at K+8 / K+24
  const int bkoff = lh * 8;    // B: lanes 16-31 start at K+8 / K+24
  f32x8 c00{}, c01{}, c10{}, c11{};

  constexpr int NIT = DIM / KSTEP;
  for (int it = 0; it < NIT; ++it) {
    const int cur = it & 1;
    if (wave < 2) __builtin_amdgcn_s_wait_tensorcnt(0);
    __syncthreads();
    if ((it + 1) < NIT) {
      const int k1 = (it + 1) * KSTEP;
      if (wave == 0) tdm_load_tile(lds_off(&lA[cur ^ 1][0][0]), gA + k1, 128);
      if (wave == 1) tdm_load_tile(lds_off(&lB[cur ^ 1][0][0]), gW + k1, 64);
    }

#pragma unroll
    for (int kc = 0; kc < 2; ++kc) {         // two 32-wide WMMA K-chunks
      const int ka = kc * 32 + akoff;
      const int kb = kc * 32 + bkoff;
      BF16Frag a0, a1, b0, b1;
      a0.h[0] = *(const bf16x8*)&lA[cur][mt + lm][ka];
      a0.h[1] = *(const bf16x8*)&lA[cur][mt + lm][ka + 16];
      a1.h[0] = *(const bf16x8*)&lA[cur][mt + 16 + lm][ka];
      a1.h[1] = *(const bf16x8*)&lA[cur][mt + 16 + lm][ka + 16];
      b0.h[0] = *(const bf16x8*)&lB[cur][nt + lm][kb];
      b0.h[1] = *(const bf16x8*)&lB[cur][nt + lm][kb + 16];
      b1.h[0] = *(const bf16x8*)&lB[cur][nt + 16 + lm][kb];
      b1.h[1] = *(const bf16x8*)&lB[cur][nt + 16 + lm][kb + 16];

      c00 = wmma_bf16(a0.v, b0.v, c00);
      c01 = wmma_bf16(a0.v, b1.v, c01);
      c10 = wmma_bf16(a1.v, b0.v, c10);
      c11 = wmma_bf16(a1.v, b1.v, c11);
    }
  }

#pragma unroll
  for (int r = 0; r < 8; ++r) {
    const int row0 = m0 + mt + lh * 8 + r;   // C layout: VGPR r -> rows r / r+8
    const int row1 = row0 + 16;
    const int col0 = n0 + nt + lm;
    const int col1 = col0 + 16;
    if (MODE == 0) {
      outb[(size_t)row0 * DIM + col0] = (bf16_t)c00[r];
      outb[(size_t)row0 * DIM + col1] = (bf16_t)c01[r];
      outb[(size_t)row1 * DIM + col0] = (bf16_t)c10[r];
      outb[(size_t)row1 * DIM + col1] = (bf16_t)c11[r];
    } else if (MODE == 1) {
      const int b0i = row0 >> 10, t0 = row0 & (SEQ - 1);
      const int b1i = row1 >> 10, t1 = row1 & (SEQ - 1);
      outb[((size_t)b0i * DIM + col0) * SEQ + t0] = (bf16_t)c00[r];
      outb[((size_t)b0i * DIM + col1) * SEQ + t0] = (bf16_t)c01[r];
      outb[((size_t)b1i * DIM + col0) * SEQ + t1] = (bf16_t)c10[r];
      outb[((size_t)b1i * DIM + col1) * SEQ + t1] = (bf16_t)c11[r];
    } else {
      outf[(size_t)row0 * DIM + col0] = c00[r];
      outf[(size_t)row0 * DIM + col1] = c01[r];
      outf[(size_t)row1 * DIM + col0] = c10[r];
      outf[(size_t)row1 * DIM + col1] = c11[r];
    }
  }
}

// ---------------------------------------------------------------------------
// Causal flash attention: one (b,head) + 64 query rows per workgroup.
// 4 waves, each owns 16 query rows. Online softmax in fp32.
// Qb,Kb: [B*T, D] bf16 row-major.  Vt: [B, D, T] bf16.  Ob: [B*T, D] bf16.
// ---------------------------------------------------------------------------
__global__ __launch_bounds__(128) void attn_fwd(const bf16_t* __restrict__ Qb,
                                                const bf16_t* __restrict__ Kb,
                                                const bf16_t* __restrict__ Vt,
                                                bf16_t* __restrict__ Ob) {
  __shared__ __align__(16) bf16_t plds[4][16][72];   // wave-private P staging

  const int lane = threadIdx.x & 31;
  const int wave = threadIdx.x >> 5;
  const int lm   = lane & 15;
  const int lh   = lane >> 4;

  const int qblk = blockIdx.x;            // 0..15 (64-row query blocks)
  const int bh   = blockIdx.y;            // 0..127
  const int b    = bh >> 4;
  const int head = bh & 15;
  const int h0   = head * HDIM;

  const int qrow0 = qblk * 64 + wave * 16;   // local row within SEQ
  const int akoff = lh * 8;
  const int bkoff = lh * 8;

  // Hoisted Q A-fragments over the 64-wide head dim (2 k-steps of 32)
  const bf16_t* qp = Qb + (size_t)(b * SEQ + qrow0 + lm) * DIM + h0;
  BF16Frag aq0, aq1;
  aq0.h[0] = *(const bf16x8*)(qp + akoff);
  aq0.h[1] = *(const bf16x8*)(qp + akoff + 16);
  aq1.h[0] = *(const bf16x8*)(qp + 32 + akoff);
  aq1.h[1] = *(const bf16x8*)(qp + 32 + akoff + 16);

  float mrun[8], lrun[8];
  f32x8 oacc[4];
#pragma unroll
  for (int r = 0; r < 8; ++r) { mrun[r] = -__builtin_inff(); lrun[r] = 0.f; }
#pragma unroll
  for (int hs = 0; hs < 4; ++hs) oacc[hs] = f32x8{};

  for (int j = 0; j <= qblk; ++j) {
    // ---- S = (Q K^T) * 1/sqrt(H), causal mask on diagonal block ----
    f32x8 s[4];
    const bf16_t* kp = Kb + (size_t)(b * SEQ + j * 64 + lm) * DIM + h0;
    if (j + 1 <= qblk)
      __builtin_prefetch(kp + (size_t)64 * DIM, 0, 0);
#pragma unroll
    for (int ns = 0; ns < 4; ++ns) {
      const bf16_t* kr = kp + (size_t)(ns * 16) * DIM;
      BF16Frag bk0, bk1;
      bk0.h[0] = *(const bf16x8*)(kr + bkoff);
      bk0.h[1] = *(const bf16x8*)(kr + bkoff + 16);
      bk1.h[0] = *(const bf16x8*)(kr + 32 + bkoff);
      bk1.h[1] = *(const bf16x8*)(kr + 32 + bkoff + 16);
      f32x8 acc{};
      acc = wmma_bf16(aq0.v, bk0.v, acc);
      acc = wmma_bf16(aq1.v, bk1.v, acc);
      s[ns] = acc;
    }

    const bool diag = (j == qblk);
#pragma unroll
    for (int ns = 0; ns < 4; ++ns) {
#pragma unroll
      for (int r = 0; r < 8; ++r) {
        float v = s[ns][r] * 0.125f;     // 1/sqrt(64)
        if (diag) {
          const int qi = qrow0 + lh * 8 + r;
          const int ki = j * 64 + ns * 16 + lm;
          if (ki > qi) v = -__builtin_inff();
        }
        s[ns][r] = v;
      }
    }

    // ---- online softmax; rows live in 16-lane halves, reduce with shfl_xor ----
#pragma unroll
    for (int r = 0; r < 8; ++r) {
      float mx = fmaxf(fmaxf(s[0][r], s[1][r]), fmaxf(s[2][r], s[3][r]));
#pragma unroll
      for (int off = 1; off < 16; off <<= 1)
        mx = fmaxf(mx, __shfl_xor(mx, off, 32));
      const float mtot  = fmaxf(mrun[r], mx);
      const float alpha = __expf(mrun[r] - mtot);
      float rs = 0.f;
#pragma unroll
      for (int ns = 0; ns < 4; ++ns) {
        const float p = __expf(s[ns][r] - mtot);
        s[ns][r] = p;
        rs += p;
      }
#pragma unroll
      for (int off = 1; off < 16; off <<= 1)
        rs += __shfl_xor(rs, off, 32);
      lrun[r] = lrun[r] * alpha + rs;
      mrun[r] = mtot;
#pragma unroll
      for (int hs = 0; hs < 4; ++hs) oacc[hs][r] *= alpha;
    }

    // ---- P: C-layout scatter to LDS, reload in A-fragment layout ----
#pragma unroll
    for (int ns = 0; ns < 4; ++ns)
#pragma unroll
      for (int r = 0; r < 8; ++r)
        plds[wave][lh * 8 + r][ns * 16 + lm] = (bf16_t)s[ns][r];

    BF16Frag ap0, ap1;
    ap0.h[0] = *(const bf16x8*)&plds[wave][lm][akoff];
    ap0.h[1] = *(const bf16x8*)&plds[wave][lm][akoff + 16];
    ap1.h[0] = *(const bf16x8*)&plds[wave][lm][32 + akoff];
    ap1.h[1] = *(const bf16x8*)&plds[wave][lm][32 + akoff + 16];

    // ---- O += P @ V (V^T rows are contiguous along t) ----
    const bf16_t* vp = Vt + ((size_t)b * DIM + h0 + lm) * SEQ + j * 64;
#pragma unroll
    for (int hs = 0; hs < 4; ++hs) {
      const bf16_t* vr = vp + (size_t)(hs * 16) * SEQ;
      BF16Frag bv0, bv1;
      bv0.h[0] = *(const bf16x8*)(vr + bkoff);
      bv0.h[1] = *(const bf16x8*)(vr + bkoff + 16);
      bv1.h[0] = *(const bf16x8*)(vr + 32 + bkoff);
      bv1.h[1] = *(const bf16x8*)(vr + 32 + bkoff + 16);
      oacc[hs] = wmma_bf16(ap0.v, bv0.v, oacc[hs]);
      oacc[hs] = wmma_bf16(ap1.v, bv1.v, oacc[hs]);
    }
  }

  // ---- normalize and store O as bf16 ----
#pragma unroll
  for (int r = 0; r < 8; ++r) {
    const float inv = 1.0f / lrun[r];
    const int row = b * SEQ + qrow0 + lh * 8 + r;
#pragma unroll
    for (int hs = 0; hs < 4; ++hs)
      Ob[(size_t)row * DIM + h0 + hs * 16 + lm] = (bf16_t)(oacc[hs][r] * inv);
  }
}

// ---------------------------------------------------------------------------
extern "C" void kernel_launch(void* const* d_in, const int* in_sizes, int n_in,
                              void* d_out, int out_size, void* d_ws, size_t ws_size,
                              hipStream_t stream) {
  const float* x  = (const float*)d_in[0];
  const float* Wk = (const float*)d_in[1];
  const float* Wq = (const float*)d_in[2];
  const float* Wv = (const float*)d_in[3];
  const float* Wo = (const float*)d_in[4];
  float* out = (float*)d_out;

  const size_t nX = (size_t)MTOT * DIM;   // 8M elements
  const size_t nW = (size_t)DIM * DIM;    // 1M elements

  // Workspace layout (bf16): xb | wqb | wkb | wvb | wob | qb | kb | vt  (~72 MB)
  bf16_t* xb  = (bf16_t*)d_ws;
  bf16_t* wqb = xb  + nX;
  bf16_t* wkb = wqb + nW;
  bf16_t* wvb = wkb + nW;
  bf16_t* wob = wvb + nW;
  bf16_t* qb  = wob + nW;
  bf16_t* kb  = qb  + nX;
  bf16_t* vt  = kb  + nX;
  bf16_t* ob  = xb;   // xb is dead after the three projection GEMMs

  cvt_f32_to_bf16<<<(int)(nX / (256 * 8)), 256, 0, stream>>>(x,  xb,  (int)nX);
  cvt_f32_to_bf16<<<(int)(nW / (256 * 8)), 256, 0, stream>>>(Wq, wqb, (int)nW);
  cvt_f32_to_bf16<<<(int)(nW / (256 * 8)), 256, 0, stream>>>(Wk, wkb, (int)nW);
  cvt_f32_to_bf16<<<(int)(nW / (256 * 8)), 256, 0, stream>>>(Wv, wvb, (int)nW);
  cvt_f32_to_bf16<<<(int)(nW / (256 * 8)), 256, 0, stream>>>(Wo, wob, (int)nW);

  dim3 ggrid(MTOT / 128, DIM / 64);
  gemm_xwt<0><<<ggrid, 256, 0, stream>>>(xb, wqb, qb, nullptr);
  gemm_xwt<0><<<ggrid, 256, 0, stream>>>(xb, wkb, kb, nullptr);
  gemm_xwt<1><<<ggrid, 256, 0, stream>>>(xb, wvb, vt, nullptr);

  dim3 agrid(SEQ / 64, BATCH * NHEAD);
  attn_fwd<<<agrid, 128, 0, stream>>>(qb, kb, vt, ob);

  gemm_xwt<2><<<ggrid, 256, 0, stream>>>(ob, wob, nullptr, out);
}
